// T4c22GNN_74388833567157
// MI455X (gfx1250) — matrix operand order repaired and checked by reading, not compiled, not run on hardware
//
#include <hip/hip_runtime.h>

// ---------------------------------------------------------------------------
// T4c22 GNN on MI455X (gfx1250, wave32, WMMA)
//
// Edge GEMMs ([E,64]@[64,32] x3 layers, [E,32]@[32,64] final) dominate
// (~27 GFLOP); node tables (6.4MB) live in the 192MB L2, so per-edge gathers
// are L2 hits -> compute-bound -> f16 WMMA 16x16x32 w/ f32 accumulation on
// 16-edge tiles per wave. Cross-lane LayerNorm reductions use DPP16 row_ror
// (pure VALU) instead of ds_bpermute to kill the serialized DS-wait chains
// seen in round 1.
// ---------------------------------------------------------------------------

typedef __attribute__((ext_vector_type(16))) _Float16 v16h;
typedef __attribute__((ext_vector_type(8)))  float    v8f;

#define LN_EPS 1e-5f

// dims fixed by the reference model
#define DIN  16
#define H0   64
#define DD   32
#define HF   64
#define OUTC 3

__device__ __forceinline__ float gelu_exact(float x) {
    return 0.5f * x * (1.0f + erff(x * 0.7071067811865475f));
}

// v + (v rotated by CTRL within the 16-lane row): DPP16, no LDS.
template <int CTRL>
__device__ __forceinline__ float dpp_add(float v) {
    int r = __builtin_amdgcn_update_dpp(0, __builtin_bit_cast(int, v),
                                        CTRL, 0xf, 0xf, true);
    return v + __builtin_bit_cast(float, r);
}
// full 16-lane row sum, result in every lane of the row (rows 0-15 / 16-31
// are independent, matching the WMMA C-layout half-wave groups).
__device__ __forceinline__ float red16_dpp(float v) {
    v = dpp_add<0x128>(v);   // row_ror:8
    v = dpp_add<0x124>(v);   // row_ror:4
    v = dpp_add<0x122>(v);   // row_ror:2
    v = dpp_add<0x121>(v);   // row_ror:1
    return v;
}

// ---------------------------------------------------------------------------
// Node embedding MLP: x[N,16] -> gelu(bn(.@W1)) [N,64] -> gelu(bn(.@W2)) [N,32]
// Scalar per node (0.15 GMAC total), weights in LDS.
// ---------------------------------------------------------------------------
__global__ void __launch_bounds__(256)
emb_kernel(const float* __restrict__ x,
           const float* __restrict__ W1, const float* __restrict__ b1,
           const float* __restrict__ g1, const float* __restrict__ bb1,
           const float* __restrict__ W2, const float* __restrict__ b2,
           const float* __restrict__ g2, const float* __restrict__ bb2,
           float* __restrict__ h, float* __restrict__ pre,
           _Float16* __restrict__ hhalf, int N)
{
    __shared__ float sW1[DIN * H0];
    __shared__ float sW2[H0 * DD];
    __shared__ float sb1[H0], sg1[H0], sB1[H0];
    __shared__ float sb2[DD], sg2[DD], sB2[DD];
    const float bnscale = 1.0f / sqrtf(1.0f + LN_EPS);
    for (int i = threadIdx.x; i < DIN * H0; i += blockDim.x) sW1[i] = W1[i];
    for (int i = threadIdx.x; i < H0 * DD; i += blockDim.x) sW2[i] = W2[i];
    for (int i = threadIdx.x; i < H0; i += blockDim.x) {
        sb1[i] = b1[i]; sg1[i] = g1[i] * bnscale; sB1[i] = bb1[i];
    }
    for (int i = threadIdx.x; i < DD; i += blockDim.x) {
        sb2[i] = b2[i]; sg2[i] = g2[i] * bnscale; sB2[i] = bb2[i];
    }
    __syncthreads();

    int n = blockIdx.x * blockDim.x + threadIdx.x;
    if (n >= N) return;

    float xin[DIN];
#pragma unroll
    for (int k = 0; k < DIN; ++k) xin[k] = x[n * DIN + k];

    float t[H0];
#pragma unroll
    for (int j = 0; j < H0; ++j) {
        float a = sb1[j];
#pragma unroll
        for (int k = 0; k < DIN; ++k) a += xin[k] * sW1[k * H0 + j];
        t[j] = gelu_exact(a * sg1[j] + sB1[j]);
    }
#pragma unroll
    for (int o = 0; o < DD; ++o) {
        float a = sb2[o];
#pragma unroll
        for (int j = 0; j < H0; ++j) a += t[j] * sW2[j * DD + o];
        float v = gelu_exact(a * sg2[o] + sB2[o]);
        h[(size_t)n * DD + o]   = v;
        pre[(size_t)n * DD + o] = v;
        hhalf[(size_t)n * DD + o] = (_Float16)v;
    }
}

// ---------------------------------------------------------------------------
// Message + scatter-aggregate kernel (the hot loop, WMMA).
// One wave = one 16-edge tile per iteration.
//   A (16x64 f16): rows = concat(h[dst], h[src]); two 16x16x32 K-slabs
//   B (64x32 f16): msg_W, staged into 4 register operand tiles per wave
//   C (16x32 f32): +bias -> LayerNorm(32) via DPP row reductions -> GELU
//                   -> atomicAdd into agg[dst]  (agg lives in L2)
// No DS instructions in the loop: reductions are DPP, per-row dst indices
// come from v_readlane (row lane ids are wave-uniform) + select.
// ---------------------------------------------------------------------------
__global__ void __launch_bounds__(256)
msg_kernel(const int* __restrict__ ei, const _Float16* __restrict__ hhalf,
           const float* __restrict__ W, const float* __restrict__ bias,
           const float* __restrict__ lng, const float* __restrict__ lnb,
           float* __restrict__ agg, int E, int ntiles)
{
    const int lane = threadIdx.x & 31;
    const int nlo  = lane & 15;
    const int hi   = lane >> 4;     // half-wave group
    const int k0   = hi * 8;        // A-operand K base for this group

    // B operand tiles: B[kblock][ntile]; lane holds col n=nlo,
    // group holds K = k0..k0+7 and k0+16..k0+23 within the 32-K slab.
    v16h B[2][2];
#pragma unroll
    for (int kb = 0; kb < 2; ++kb) {
#pragma unroll
        for (int nt = 0; nt < 2; ++nt) {
            v16h bv;
#pragma unroll
            for (int i = 0; i < 16; ++i) {
                int k = kb * 32 + k0 + (i & 7) + ((i >> 3) << 4);
                bv[i] = (_Float16)W[k * DD + nt * 16 + nlo];
            }
            B[kb][nt] = bv;
        }
    }
    const float mb0 = bias[nlo],     mb1 = bias[16 + nlo];
    const float g0  = lng[nlo],      g1  = lng[16 + nlo];
    const float be0 = lnb[nlo],      be1 = lnb[16 + nlo];

    const int wave = (int)((blockIdx.x * blockDim.x + threadIdx.x) >> 5);
    const int nw   = (int)((gridDim.x * blockDim.x) >> 5);

    for (int tile = wave; tile < ntiles; tile += nw) {
        int e = tile * 16 + nlo;            // lanes l and l+16 share edge l
        if (e >= E) e = E - 1;              // clamp (keeps EXEC full)
        const int s = ei[e];                // x_j source
        const int d = ei[E + e];            // x_i target / aggregation index

        const _Float16* rd = hhalf + (size_t)d * DD;
        const _Float16* rs = hhalf + (size_t)s * DD;
        union { v16h v; struct { uint4 lo; uint4 hi; } q; } a0, a1;
        // A layout: group lo holds K=k0..k0+7 and k0+16..k0+23 of its row
        a0.q.lo = *(const uint4*)(rd + k0);
        a0.q.hi = *(const uint4*)(rd + k0 + 16);
        a1.q.lo = *(const uint4*)(rs + k0);
        a1.q.hi = *(const uint4*)(rs + k0 + 16);

        v8f c0 = {}; v8f c1 = {};
        c0 = __builtin_amdgcn_wmma_f32_16x16x32_f16(false, a0.v, false, B[0][0], (short)0, c0, false, false);
        c0 = __builtin_amdgcn_wmma_f32_16x16x32_f16(false, a1.v, false, B[1][0], (short)0, c0, false, false);
        c1 = __builtin_amdgcn_wmma_f32_16x16x32_f16(false, a0.v, false, B[0][1], (short)0, c1, false, false);
        c1 = __builtin_amdgcn_wmma_f32_16x16x32_f16(false, a1.v, false, B[1][1], (short)0, c1, false, false);

        // C layout: reg r holds row (r + 8*hi), lane holds col nlo / 16+nlo.
        float x0[8], x1[8], sm[8], sq[8];
#pragma unroll
        for (int r = 0; r < 8; ++r) {
            x0[r] = c0[r] + mb0;
            x1[r] = c1[r] + mb1;
            sm[r] = x0[r] + x1[r];
            sq[r] = x0[r] * x0[r] + x1[r] * x1[r];
        }
        // 16 independent DPP reduction chains, no memory waits
#pragma unroll
        for (int r = 0; r < 8; ++r) {
            sm[r] = red16_dpp(sm[r]);
            sq[r] = red16_dpp(sq[r]);
        }
#pragma unroll
        for (int r = 0; r < 8; ++r) {
            float mean = sm[r] * (1.0f / 32.0f);
            float var  = sq[r] * (1.0f / 32.0f) - mean * mean;
            float inv  = rsqrtf(var + LN_EPS);
            float m0 = gelu_exact((x0[r] - mean) * inv * g0 + be0);
            float m1 = gelu_exact((x1[r] - mean) * inv * g1 + be1);

            // dst index of rows r / r+8: wave-uniform lane reads + select
            int dlo = __builtin_amdgcn_readlane(d, r);
            int dhi = __builtin_amdgcn_readlane(d, r + 8);
            int drow = hi ? dhi : dlo;
            int mrow = r + hi * 8;
            if (tile * 16 + mrow < E) {
                atomicAdd(&agg[(size_t)drow * DD + nlo],      m0);
                atomicAdd(&agg[(size_t)drow * DD + 16 + nlo], m1);
            }
        }
    }
}

// ---------------------------------------------------------------------------
// Node update: h += gelu(ln(concat(h, agg) @ upd_W + b)). Scalar per node.
// ---------------------------------------------------------------------------
__global__ void __launch_bounds__(256)
upd_kernel(const float* __restrict__ W, const float* __restrict__ b,
           const float* __restrict__ lg, const float* __restrict__ lb,
           float* __restrict__ h, const float* __restrict__ agg,
           _Float16* __restrict__ hhalf, int N)
{
    __shared__ float sW[2 * DD * DD];
    __shared__ float sb[DD], sg[DD], sB[DD];
    for (int i = threadIdx.x; i < 2 * DD * DD; i += blockDim.x) sW[i] = W[i];
    for (int i = threadIdx.x; i < DD; i += blockDim.x) {
        sb[i] = b[i]; sg[i] = lg[i]; sB[i] = lb[i];
    }
    __syncthreads();

    int n = blockIdx.x * blockDim.x + threadIdx.x;
    if (n >= N) return;

    float hv[DD], av[DD];
#pragma unroll
    for (int k = 0; k < DD; ++k) {
        hv[k] = h[(size_t)n * DD + k];
        av[k] = agg[(size_t)n * DD + k];
    }
    float y[DD];
    float s = 0.0f, ss = 0.0f;
#pragma unroll
    for (int o = 0; o < DD; ++o) {
        float a = sb[o];
#pragma unroll
        for (int k = 0; k < DD; ++k)
            a += hv[k] * sW[k * DD + o] + av[k] * sW[(DD + k) * DD + o];
        y[o] = a; s += a; ss += a * a;
    }
    float mean = s * (1.0f / DD);
    float var  = ss * (1.0f / DD) - mean * mean;
    float inv  = rsqrtf(var + LN_EPS);
#pragma unroll
    for (int o = 0; o < DD; ++o) {
        float v  = (y[o] - mean) * inv * sg[o] + sB[o];
        float hn = hv[o] + gelu_exact(v);
        h[(size_t)n * DD + o] = hn;
        hhalf[(size_t)n * DD + o] = (_Float16)hn;
    }
}

// hfin = h + pre (residual around the GNN stack)
__global__ void __launch_bounds__(256)
hfin_kernel(const float* __restrict__ h, const float* __restrict__ pre,
            float* __restrict__ hfin, int total)
{
    int i = blockIdx.x * blockDim.x + threadIdx.x;
    if (i < total) hfin[i] = h[i] + pre[i];
}

// ---------------------------------------------------------------------------
// Final edge head: (hfin[dst]-hfin[src]) @ W1 (32->64), bn, gelu, @ W2 (64->3)
// WMMA for the 32->64 GEMM; 64->3 via DPP cross-lane reductions.
// ---------------------------------------------------------------------------
__global__ void __launch_bounds__(256)
final_kernel(const int* __restrict__ ei, const float* __restrict__ hfin,
             const float* __restrict__ W1, const float* __restrict__ b1,
             const float* __restrict__ bng, const float* __restrict__ bnb,
             const float* __restrict__ W2, const float* __restrict__ b2,
             float* __restrict__ out, int E, int ntiles)
{
    const int lane = threadIdx.x & 31;
    const int nlo  = lane & 15;
    const int hi   = lane >> 4;
    const int k0   = hi * 8;
    const float bnscale = 1.0f / sqrtf(1.0f + LN_EPS);

    v16h  B[4];
    float fb1[4], fg[4], fB[4], w2l[4][3];
#pragma unroll
    for (int nt = 0; nt < 4; ++nt) {
        v16h bv;
#pragma unroll
        for (int i = 0; i < 16; ++i) {
            int k = k0 + (i & 7) + ((i >> 3) << 4); // K = k0..k0+7, k0+16..k0+23
            bv[i] = (_Float16)W1[k * HF + nt * 16 + nlo];
        }
        B[nt] = bv;
        int col = nt * 16 + nlo;
        fb1[nt] = b1[col];
        fg[nt]  = bng[col] * bnscale;
        fB[nt]  = bnb[col];
#pragma unroll
        for (int o = 0; o < OUTC; ++o) w2l[nt][o] = W2[col * OUTC + o];
    }
    const float fb2[3] = { b2[0], b2[1], b2[2] };

    const int wave = (int)((blockIdx.x * blockDim.x + threadIdx.x) >> 5);
    const int nw   = (int)((gridDim.x * blockDim.x) >> 5);

    for (int tile = wave; tile < ntiles; tile += nw) {
        int e = tile * 16 + nlo;
        if (e >= E) e = E - 1;
        const int s = ei[e];
        const int d = ei[E + e];
        const float4* rd4 = (const float4*)(hfin + (size_t)d * DD);
        const float4* rs4 = (const float4*)(hfin + (size_t)s * DD);

        // A operand: K = k0..k0+7 and k0+16..k0+23 of (hfin[d]-hfin[s]),
        // gathered as 8x b128 loads.
        const int q0 = k0 >> 2;                  // 0 or 2
        float4 da = rd4[q0],     db = rd4[q0 + 1];
        float4 dc = rd4[q0 + 4], dd = rd4[q0 + 5];
        float4 sa = rs4[q0],     sb = rs4[q0 + 1];
        float4 sc = rs4[q0 + 4], sd = rs4[q0 + 5];
        v16h a;
        a[0]  = (_Float16)(da.x - sa.x); a[1]  = (_Float16)(da.y - sa.y);
        a[2]  = (_Float16)(da.z - sa.z); a[3]  = (_Float16)(da.w - sa.w);
        a[4]  = (_Float16)(db.x - sb.x); a[5]  = (_Float16)(db.y - sb.y);
        a[6]  = (_Float16)(db.z - sb.z); a[7]  = (_Float16)(db.w - sb.w);
        a[8]  = (_Float16)(dc.x - sc.x); a[9]  = (_Float16)(dc.y - sc.y);
        a[10] = (_Float16)(dc.z - sc.z); a[11] = (_Float16)(dc.w - sc.w);
        a[12] = (_Float16)(dd.x - sd.x); a[13] = (_Float16)(dd.y - sd.y);
        a[14] = (_Float16)(dd.z - sd.z); a[15] = (_Float16)(dd.w - sd.w);

        float t[4][8];
#pragma unroll
        for (int nt = 0; nt < 4; ++nt) {
            v8f c = {};
            c = __builtin_amdgcn_wmma_f32_16x16x32_f16(false, a, false, B[nt], (short)0, c, false, false);
#pragma unroll
            for (int r = 0; r < 8; ++r)
                t[nt][r] = gelu_exact((c[r] + fb1[nt]) * fg[nt] + fB[nt]);
        }

        // 64->3 projection: 24 independent DPP reduction chains
        float p[8][OUTC];
#pragma unroll
        for (int r = 0; r < 8; ++r)
#pragma unroll
            for (int o = 0; o < OUTC; ++o)
                p[r][o] = red16_dpp(t[0][r] * w2l[0][o] + t[1][r] * w2l[1][o]
                                  + t[2][r] * w2l[2][o] + t[3][r] * w2l[3][o]);
#pragma unroll
        for (int r = 0; r < 8; ++r) {
            int erow = tile * 16 + r + hi * 8;
#pragma unroll
            for (int o = 0; o < OUTC; ++o)
                if (nlo == o && erow < E)
                    out[(size_t)erow * OUTC + o] = p[r][o] + fb2[o];
        }
    }
}

// ---------------------------------------------------------------------------
extern "C" void kernel_launch(void* const* d_in, const int* in_sizes, int n_in,
                              void* d_out, int out_size, void* d_ws, size_t ws_size,
                              hipStream_t stream)
{
    const float* x        = (const float*)d_in[0];
    const int*   ei       = (const int*)  d_in[1];
    const float* emb_W1   = (const float*)d_in[2];
    const float* emb_b1   = (const float*)d_in[3];
    const float* emb_g1   = (const float*)d_in[4];
    const float* emb_bb1  = (const float*)d_in[5];
    const float* emb_W2   = (const float*)d_in[6];
    const float* emb_b2   = (const float*)d_in[7];
    const float* emb_g2   = (const float*)d_in[8];
    const float* emb_bb2  = (const float*)d_in[9];
    const float* msg_W    = (const float*)d_in[10];
    const float* msg_b    = (const float*)d_in[11];
    const float* msg_lg   = (const float*)d_in[12];
    const float* msg_lb   = (const float*)d_in[13];
    const float* upd_W    = (const float*)d_in[14];
    const float* upd_b    = (const float*)d_in[15];
    const float* upd_lg   = (const float*)d_in[16];
    const float* upd_lb   = (const float*)d_in[17];
    const float* fin_W1   = (const float*)d_in[18];
    const float* fin_b1   = (const float*)d_in[19];
    const float* fin_g    = (const float*)d_in[20];
    const float* fin_bb   = (const float*)d_in[21];
    const float* fin_W2   = (const float*)d_in[22];
    const float* fin_b2   = (const float*)d_in[23];

    const int N = in_sizes[0] / DIN;
    const int E = in_sizes[1] / 2;
    const int L = in_sizes[10] / (2 * DD * DD);

    // workspace carve-up (256B aligned regions)
    char*  ws  = (char*)d_ws;
    size_t off = 0;
    auto carve = [&](size_t bytes) -> void* {
        void* p = ws + off;
        off += (bytes + 255) & ~(size_t)255;
        return p;
    };
    float*    h     = (float*)   carve((size_t)N * DD * sizeof(float));
    float*    pre   = (float*)   carve((size_t)N * DD * sizeof(float));
    float*    agg   = (float*)   carve((size_t)N * DD * sizeof(float));
    float*    hfin  = (float*)   carve((size_t)N * DD * sizeof(float));
    _Float16* hhalf = (_Float16*)carve((size_t)N * DD * sizeof(_Float16));
    (void)ws_size;

    const int tb = 256;
    const int nblk = (N + tb - 1) / tb;
    emb_kernel<<<nblk, tb, 0, stream>>>(x, emb_W1, emb_b1, emb_g1, emb_bb1,
                                        emb_W2, emb_b2, emb_g2, emb_bb2,
                                        h, pre, hhalf, N);

    const int ntiles  = (E + 15) / 16;
    int mblocks = (ntiles + 7) / 8;           // 8 waves per block
    if (mblocks > 2048) mblocks = 2048;       // grid-stride over tiles

    for (int l = 0; l < L; ++l) {
        hipMemsetAsync(agg, 0, (size_t)N * DD * sizeof(float), stream);
        msg_kernel<<<mblocks, tb, 0, stream>>>(ei, hhalf,
                                               msg_W + (size_t)l * 2 * DD * DD,
                                               msg_b + (size_t)l * DD,
                                               msg_lg + (size_t)l * DD,
                                               msg_lb + (size_t)l * DD,
                                               agg, E, ntiles);
        upd_kernel<<<nblk, tb, 0, stream>>>(upd_W + (size_t)l * 2 * DD * DD,
                                            upd_b + (size_t)l * DD,
                                            upd_lg + (size_t)l * DD,
                                            upd_lb + (size_t)l * DD,
                                            h, agg, hhalf, N);
    }

    hfin_kernel<<<(N * DD + tb - 1) / tb, tb, 0, stream>>>(h, pre, hfin, N * DD);

    final_kernel<<<mblocks, tb, 0, stream>>>(ei, hfin, fin_W1, fin_b1,
                                             fin_g, fin_bb, fin_W2, fin_b2,
                                             (float*)d_out, E, ntiles);
}
